// RoPEMultiheadSelfAttention_6940667150438
// MI455X (gfx1250) — matrix operand order repaired and checked by reading, not compiled, not run on hardware
//
#include <hip/hip_runtime.h>
#include <cstdint>

// ---------------------------------------------------------------------------
// RoPE multi-head self-attention for MI455X (gfx1250, wave32, WMMA bf16).
//   B=4, L=2048, D=1024, H=16, hd=64
// Pipeline:
//   1) pack x, Wq, Wk, Wv, Wo to bf16
//   2) Q = rope(x@Wq^T+bq)  -> [B,H,L,hd] bf16   (WMMA GEMM, fused RoPE)
//      K = rope(x@Wk^T+bk)  -> [B,H,L,hd] bf16
//      V =      x@Wv^T+bv   -> [B,H,hd,L] bf16   (stored transposed for PV)
//   3) flash attention: K/V double-buffered in LDS via async global->LDS
//      copies (ASYNCcnt), S^T = K*Q^T formulation, online softmax -> AO bf16
//   4) out = AO@Wo^T + bo -> f32
// ---------------------------------------------------------------------------

typedef __attribute__((ext_vector_type(16))) __bf16 v16bf;
typedef __attribute__((ext_vector_type(8)))  __bf16 v8bf;
typedef __attribute__((ext_vector_type(8)))  float  v8f;

#define D_MODEL  1024
#define NHEAD    16
#define HEAD_DIM 64
#define SEQ_L    2048
#define KV_BLK   32
// ln(10000)/32
#define ROPE_LN_SCALE 0.2878231366242558f

// ---------------------------------------------------------------------------
// f32 -> bf16 pack
// ---------------------------------------------------------------------------
__global__ void pack_bf16_kernel(const float* __restrict__ in,
                                 __bf16* __restrict__ out, int n) {
  int i = blockIdx.x * blockDim.x + threadIdx.x;
  int stride = gridDim.x * blockDim.x;
  for (; i < n; i += stride) out[i] = (__bf16)in[i];
}

// ---------------------------------------------------------------------------
// Generic WMMA GEMM: out = A[M,K] @ W[N,K]^T + bias[N]
// wave computes a 32(M) x 64(N) tile (2 A-frags x 4 B-frags -> 8 WMMAs/k-step
// so each B fragment feeds two WMMAs); 8 waves/block -> 256 x 64 block tile.
// Epilogues: 0 = RoPE + store bf16 [B,H,L,hd]
//            1 = store bf16 transposed [B,H,hd,L]
//            2 = store f32 row-major [M,N]
// ---------------------------------------------------------------------------
enum { EPI_ROPE = 0, EPI_VT = 1, EPI_F32 = 2 };

template <int EPI>
__global__ __launch_bounds__(256)
void gemm_wmma_kernel(const __bf16* __restrict__ A, const __bf16* __restrict__ W,
                      const float* __restrict__ bias, void* __restrict__ outp,
                      int K, int Ltot) {
  const int lane  = threadIdx.x & 31;
  const int wave  = threadIdx.x >> 5;
  const int lmod  = lane & 15;
  const int lhalf = lane >> 4;
  const int m0 = blockIdx.y * 256 + wave * 32;
  const int n0 = blockIdx.x * 64;

  v8f acc[2][4];
#pragma unroll
  for (int mi = 0; mi < 2; ++mi)
#pragma unroll
    for (int j = 0; j < 4; ++j)
#pragma unroll
      for (int v = 0; v < 8; ++v) acc[mi][j][v] = 0.0f;

  // A fragment rows (contiguous k order, same as B: lane<16 holds k
  // [k0..k0+15]; lanes>=16 hold [k0+16..k0+31]).
  const __bf16* arow[2] = {A + (size_t)(m0 + lmod) * K,
                           A + (size_t)(m0 + 16 + lmod) * K};

  for (int k0 = 0; k0 < K; k0 += 32) {
    // prefetch the K-slice a few steps ahead (weights are L2-resident)
    if (k0 + 128 < K) {
      __builtin_prefetch(arow[0] + k0 + 128, 0, 1);
      __builtin_prefetch(arow[1] + k0 + 128, 0, 1);
      __builtin_prefetch(W + (size_t)(n0 + lmod * 4) * K + k0 + 128, 0, 1);
    }
    v16bf av[2];
#pragma unroll
    for (int mi = 0; mi < 2; ++mi) {
      v8bf a0 = *(const v8bf*)(arow[mi] + k0 + lhalf * 16);
      v8bf a1 = *(const v8bf*)(arow[mi] + k0 + lhalf * 16 + 8);
      av[mi] = __builtin_shufflevector(a0, a1, 0, 1, 2, 3, 4, 5, 6, 7,
                                       8, 9, 10, 11, 12, 13, 14, 15);
    }
#pragma unroll
    for (int j = 0; j < 4; ++j) {
      // B fragment: lane<16 -> W[n][k0..k0+15], lanes>=16 -> W[n][k0+16..k0+31]
      const __bf16* wrow =
          W + (size_t)(n0 + j * 16 + lmod) * K + k0 + lhalf * 16;
      v16bf bv = *(const v16bf*)wrow;
      acc[0][j] = __builtin_amdgcn_wmma_f32_16x16x32_bf16(
          false, av[0], false, bv, (short)0, acc[0][j], false, false);
      acc[1][j] = __builtin_amdgcn_wmma_f32_16x16x32_bf16(
          false, av[1], false, bv, (short)0, acc[1][j], false, false);
    }
  }

  // bias (per-column, column fixed per lane per tile)
#pragma unroll
  for (int j = 0; j < 4; ++j) {
    float b = bias[n0 + j * 16 + lmod];
#pragma unroll
    for (int mi = 0; mi < 2; ++mi)
#pragma unroll
      for (int v = 0; v < 8; ++v) acc[mi][j][v] += b;
  }

  const int bidx = m0 / Ltot;

#pragma unroll
  for (int mi = 0; mi < 2; ++mi) {
    const int posbase = (m0 % Ltot) + mi * 16 + lhalf * 8;

    if constexpr (EPI == EPI_ROPE) {
      // C layout: lane holds column n = n0+j*16+lmod, rows posbase+v.
      // Rotate-half partner column (d^32) lives in tile j^2 of the SAME lane.
      __bf16* Out = (__bf16*)outp;
#pragma unroll
      for (int j = 0; j < 4; ++j) {
        int n = n0 + j * 16 + lmod;
        int h = n >> 6, d = n & 63;
        float sign = (d < 32) ? -1.0f : 1.0f;
        float invf = __expf(-(float)(d & 31) * ROPE_LN_SCALE);
        size_t base =
            ((size_t)(bidx * NHEAD + h) * (size_t)Ltot) * HEAD_DIM + (size_t)d;
#pragma unroll
        for (int v = 0; v < 8; ++v) {
          int pos = posbase + v;
          float sv, cv;
          __sincosf((float)pos * invf, &sv, &cv);
          float r = acc[mi][j][v] * cv + sign * acc[mi][j ^ 2][v] * sv;
          Out[base + (size_t)pos * HEAD_DIM] = (__bf16)r;
        }
      }
    } else if constexpr (EPI == EPI_VT) {
      // Store transposed [B,H,hd,L]: lane's 8 rows are consecutive positions
      // -> one contiguous 16B store per tile.
      __bf16* Out = (__bf16*)outp;
#pragma unroll
      for (int j = 0; j < 4; ++j) {
        int n = n0 + j * 16 + lmod;
        int h = n >> 6, d = n & 63;
        v8bf pk;
#pragma unroll
        for (int v = 0; v < 8; ++v) pk[v] = (__bf16)acc[mi][j][v];
        size_t base =
            ((size_t)(bidx * NHEAD + h) * HEAD_DIM + d) * (size_t)Ltot +
            (size_t)posbase;
        *(v8bf*)(Out + base) = pk;
      }
    } else {
      float* Out = (float*)outp;
#pragma unroll
      for (int j = 0; j < 4; ++j) {
        int n = n0 + j * 16 + lmod;
#pragma unroll
        for (int v = 0; v < 8; ++v) {
          int m = m0 + mi * 16 + lhalf * 8 + v;
          Out[(size_t)m * D_MODEL + n] = acc[mi][j][v];
        }
      }
    }
  }
}

// ---------------------------------------------------------------------------
// Flash attention: 4 waves/block share K/V blocks staged in LDS via
// gfx1250 async global->LDS copies (double buffered, ASYNCcnt-tracked).
// Per wave: 16 queries x full hd=64.  S^T = K(32k x 64) * Q^T(64 x 16q):
// per-query softmax stats are in-lane (16 vals) + one shfl_xor(16); the P
// fragment for the PV WMMA is built entirely in-lane (S^T C-layout == P
// A-layout).
// ---------------------------------------------------------------------------
__global__ __launch_bounds__(128)
void flash_attn_kernel(const __bf16* __restrict__ Q, const __bf16* __restrict__ Kc,
                       const __bf16* __restrict__ Vt, __bf16* __restrict__ AO,
                       int Ltot) {
  const int lane  = threadIdx.x & 31;
  const int wave  = threadIdx.x >> 5;
  const int lmod  = lane & 15;
  const int lhalf = lane >> 4;
  const int bh = blockIdx.y;
  const int b  = bh >> 4;
  const int h  = bh & 15;
  const int q0 = blockIdx.x * 64 + wave * 16;

  const __bf16* Qh = Q  + (size_t)bh * Ltot * HEAD_DIM;   // [q][d]
  const __bf16* Kh = Kc + (size_t)bh * Ltot * HEAD_DIM;   // [key][d]
  const __bf16* Vh = Vt + (size_t)bh * HEAD_DIM * Ltot;   // [d][key]

  // Double-buffered K (32 keys x 64 dims) and V^T (64 dims x 32 keys) tiles.
  __shared__ alignas(128) __bf16 sK[2][KV_BLK][HEAD_DIM];  // 8 KB
  __shared__ alignas(128) __bf16 sV[2][HEAD_DIM][KV_BLK];  // 8 KB

  // Issue async global->LDS copies for one KV block (all 128 threads).
  // Each instruction moves 16B/lane straight into LDS (no VGPR staging).
  auto issue_kv = [&](int kv0, int buf) {
#pragma unroll
    for (int u = 0; u < 2; ++u) {
      int unit = threadIdx.x + u * 128;            // 0..255 -> 4KB K tile
      int row = unit >> 3, chunk = unit & 7;
      const __bf16* src = Kh + (size_t)(kv0 + row) * HEAD_DIM + chunk * 8;
      uint32_t dst = (uint32_t)(uintptr_t)&sK[buf][row][chunk * 8];
      asm volatile("global_load_async_to_lds_b128 %0, %1, off"
                   :: "v"(dst), "v"(src) : "memory");
    }
#pragma unroll
    for (int u = 0; u < 2; ++u) {
      int unit = threadIdx.x + u * 128;            // 0..255 -> 4KB V tile
      int row = unit >> 2, chunk = unit & 3;
      const __bf16* src = Vh + (size_t)row * Ltot + kv0 + chunk * 8;
      uint32_t dst = (uint32_t)(uintptr_t)&sV[buf][row][chunk * 8];
      asm volatile("global_load_async_to_lds_b128 %0, %1, off"
                   :: "v"(dst), "v"(src) : "memory");
    }
  };

  // Q as B fragments (64 dims -> two K-steps of 32), read once from global.
  v16bf qb[2];
#pragma unroll
  for (int s = 0; s < 2; ++s)
    qb[s] = *(const v16bf*)(Qh + (size_t)(q0 + lmod) * HEAD_DIM + s * 32 +
                            lhalf * 16);

  v8f o[4];
#pragma unroll
  for (int j = 0; j < 4; ++j)
#pragma unroll
    for (int v = 0; v < 8; ++v) o[j][v] = 0.0f;

  float mrun = -1e30f, lsum = 0.0f;

  issue_kv(0, 0);
  int buf = 0;

  for (int kv0 = 0; kv0 < Ltot; kv0 += KV_BLK) {
    // Each wave drains its own ASYNCcnt, then the workgroup barrier makes the
    // whole tile's completion visible to every wave.
    asm volatile("s_wait_asynccnt 0" ::: "memory");
    __syncthreads();
    if (kv0 + KV_BLK < Ltot) issue_kv(kv0 + KV_BLK, buf ^ 1);

    // --- scores: S^T tiles (keys 0-15 / 16-31 of this block), from LDS ---
    v8f st[2];
#pragma unroll
    for (int t = 0; t < 2; ++t)
#pragma unroll
      for (int v = 0; v < 8; ++v) st[t][v] = 0.0f;

#pragma unroll
    for (int t = 0; t < 2; ++t) {
#pragma unroll
      for (int s = 0; s < 2; ++s) {
        const __bf16* kr = &sK[buf][t * 16 + lmod][s * 32];
        v8bf a0 = *(const v8bf*)(kr + lhalf * 16);
        v8bf a1 = *(const v8bf*)(kr + lhalf * 16 + 8);
        v16bf av = __builtin_shufflevector(a0, a1, 0, 1, 2, 3, 4, 5, 6, 7,
                                           8, 9, 10, 11, 12, 13, 14, 15);
        st[t] = __builtin_amdgcn_wmma_f32_16x16x32_bf16(
            false, av, false, qb[s], (short)0, st[t], false, false);
      }
    }

    // --- online softmax (per query = per column of S^T) ---
    float mloc = -1e30f;
#pragma unroll
    for (int t = 0; t < 2; ++t)
#pragma unroll
      for (int v = 0; v < 8; ++v) {
        st[t][v] *= 0.125f;  // 1/sqrt(64)
        mloc = fmaxf(mloc, st[t][v]);
      }
    mloc = fmaxf(mloc, __shfl_xor(mloc, 16, 32));
    float mnew  = fmaxf(mrun, mloc);
    float alpha = __expf(mrun - mnew);

    float psum = 0.0f;
    v16bf pf;
#pragma unroll
    for (int t = 0; t < 2; ++t)
#pragma unroll
      for (int v = 0; v < 8; ++v) {
        float p = __expf(st[t][v] - mnew);
        psum += p;
        pf[t * 8 + v] = (__bf16)p;  // matches A-layout of P (16q x 32k)
      }
    psum += __shfl_xor(psum, 16, 32);
    lsum = lsum * alpha + psum;
    mrun = mnew;

    // per-vgpr output rescale factor: query index = v + 8*lhalf
    float scl[8];
#pragma unroll
    for (int v = 0; v < 8; ++v) scl[v] = __shfl(alpha, v + lhalf * 8, 32);

    // --- PV: O += P(16x32) * V(32x64), V fragments from LDS ---
#pragma unroll
    for (int j = 0; j < 4; ++j) {
      v8bf b0 = *(const v8bf*)&sV[buf][j * 16 + lmod][lhalf * 8];
      v8bf b1 = *(const v8bf*)&sV[buf][j * 16 + lmod][16 + lhalf * 8];
      v16bf vb = __builtin_shufflevector(b0, b1, 0, 1, 2, 3, 4, 5, 6, 7,
                                         8, 9, 10, 11, 12, 13, 14, 15);
#pragma unroll
      for (int v = 0; v < 8; ++v) o[j][v] *= scl[v];
      o[j] = __builtin_amdgcn_wmma_f32_16x16x32_bf16(
          false, pf, false, vb, (short)0, o[j], false, false);
    }

    __syncthreads();  // all waves done reading buf before it is refilled
    buf ^= 1;
  }

  float linv[8];
#pragma unroll
  for (int v = 0; v < 8; ++v)
    linv[v] = 1.0f / __shfl(lsum, v + lhalf * 8, 32);

#pragma unroll
  for (int j = 0; j < 4; ++j) {
    int d = h * HEAD_DIM + j * 16 + lmod;
#pragma unroll
    for (int v = 0; v < 8; ++v) {
      int q = q0 + lhalf * 8 + v;
      AO[(size_t)(b * Ltot + q) * D_MODEL + d] = (__bf16)(o[j][v] * linv[v]);
    }
  }
}

// ---------------------------------------------------------------------------
extern "C" void kernel_launch(void* const* d_in, const int* in_sizes, int n_in,
                              void* d_out, int out_size, void* d_ws,
                              size_t ws_size, hipStream_t stream) {
  (void)n_in; (void)out_size; (void)ws_size;

  const float* x  = (const float*)d_in[0];
  const float* Wq = (const float*)d_in[1];
  const float* bq = (const float*)d_in[2];
  const float* Wk = (const float*)d_in[3];
  const float* bk = (const float*)d_in[4];
  const float* Wv = (const float*)d_in[5];
  const float* bv = (const float*)d_in[6];
  const float* Wo = (const float*)d_in[7];
  const float* bo = (const float*)d_in[8];

  const int Mtot  = in_sizes[0] / D_MODEL;  // B*L
  const int Batch = Mtot / SEQ_L;

  size_t off = 0;
  auto alloc = [&](size_t bytes) -> void* {
    void* p = (char*)d_ws + off;
    off += (bytes + 255) & ~(size_t)255;
    return p;
  };
  const size_t actElems = (size_t)Mtot * D_MODEL;      // B*L*D
  const size_t wElems   = (size_t)D_MODEL * D_MODEL;   // 1024*1024

  __bf16* xb  = (__bf16*)alloc(actElems * 2);
  __bf16* Wqb = (__bf16*)alloc(wElems * 2);
  __bf16* Wkb = (__bf16*)alloc(wElems * 2);
  __bf16* Wvb = (__bf16*)alloc(wElems * 2);
  __bf16* Wob = (__bf16*)alloc(wElems * 2);
  __bf16* Qb  = (__bf16*)alloc(actElems * 2);  // [B,H,L,hd]
  __bf16* Kb  = (__bf16*)alloc(actElems * 2);  // [B,H,L,hd]
  __bf16* Vtb = (__bf16*)alloc(actElems * 2);  // [B,H,hd,L]
  __bf16* AOb = (__bf16*)alloc(actElems * 2);  // [B*L, D]

  // 1) pack to bf16
  pack_bf16_kernel<<<2048, 256, 0, stream>>>(x, xb, (int)actElems);
  pack_bf16_kernel<<<1024, 256, 0, stream>>>(Wq, Wqb, (int)wElems);
  pack_bf16_kernel<<<1024, 256, 0, stream>>>(Wk, Wkb, (int)wElems);
  pack_bf16_kernel<<<1024, 256, 0, stream>>>(Wv, Wvb, (int)wElems);
  pack_bf16_kernel<<<1024, 256, 0, stream>>>(Wo, Wob, (int)wElems);

  // 2) QKV projections (WMMA) with fused RoPE / transpose epilogues
  dim3 gg(D_MODEL / 64, Mtot / 256);
  gemm_wmma_kernel<EPI_ROPE><<<gg, 256, 0, stream>>>(xb, Wqb, bq, Qb, D_MODEL, SEQ_L);
  gemm_wmma_kernel<EPI_ROPE><<<gg, 256, 0, stream>>>(xb, Wkb, bk, Kb, D_MODEL, SEQ_L);
  gemm_wmma_kernel<EPI_VT  ><<<gg, 256, 0, stream>>>(xb, Wvb, bv, Vtb, D_MODEL, SEQ_L);

  // 3) flash attention (WMMA + async LDS staging)
  dim3 ga(SEQ_L / 64, Batch * NHEAD);
  flash_attn_kernel<<<ga, 128, 0, stream>>>(Qb, Kb, Vtb, AOb, SEQ_L);

  // 4) output projection -> f32
  gemm_wmma_kernel<EPI_F32><<<gg, 256, 0, stream>>>(AOb, Wob, bo, (float*)d_out,
                                                    D_MODEL, SEQ_L);
}
